// Seq2Seq_65859028517201
// MI455X (gfx1250) — matrix operand order repaired and checked by reading, not compiled, not run on hardware
//
#include <hip/hip_runtime.h>
#include <hip/hip_bf16.h>
#include <stdint.h>
#include <stddef.h>

// Problem constants (from reference): H=1024, V=30000, NT=256, B=128, T=50
#define H_    1024
#define H2_   (2 * H_)
#define H3_   (3 * H_)
#define V_    30000
#define NT_   256
#define B_    128
#define T_    50
#define OID_  1

typedef __bf16 bf16;
typedef __attribute__((ext_vector_type(8)))  bf16  v8bf;
typedef __attribute__((ext_vector_type(16))) bf16  v16bf;
typedef __attribute__((ext_vector_type(8)))  float v8f;

__device__ __forceinline__ float sigmoid_(float x) { return 1.0f / (1.0f + __expf(-x)); }

// ---------------------------------------------------------------------------
// Utility kernels
// ---------------------------------------------------------------------------
__global__ void k_cast_bf16(const float* __restrict__ s, bf16* __restrict__ d, size_t n) {
  for (size_t i = (size_t)blockIdx.x * blockDim.x + threadIdx.x; i < n;
       i += (size_t)gridDim.x * blockDim.x)
    d[i] = (bf16)s[i];
}

__global__ void k_zero_u32(uint32_t* __restrict__ p, size_t n) {
  for (size_t i = (size_t)blockIdx.x * blockDim.x + threadIdx.x; i < n;
       i += (size_t)gridDim.x * blockDim.x)
    p[i] = 0u;
}

// ---------------------------------------------------------------------------
// WMMA GEMM: C[M,N] = A[M,K](bf16, row-major, lda) @ W[N,K]^T(bf16, row-major)
//            + bias; op: 0 = identity, 1 = tanh; optional bf16 mirror of C.
// NB = N-tiles per wave: one A fragment is reused across NB B fragments,
// cutting A traffic NB-fold and raising wmma density in the inner loop.
// grid = (N/(16*NB), M/128); block = 256 = 8 wave32, wave w owns M-tile w.
// Fragment addressing (contiguous k order, shared by A and B):
//   A 16x32: lanes<16 rows M=l, K [0..15]; lanes>=16 K [16..31].
//   B 32x16: lanes<16 col N=l, K [0..15]; lanes>=16 K [16..31].
// ---------------------------------------------------------------------------
template <int NB>
__global__ __launch_bounds__(256) void k_gemm_bf16(
    const bf16* __restrict__ A, int lda,
    const bf16* __restrict__ W,
    const float* __restrict__ bias,
    float* __restrict__ C, long ldc,
    bf16* __restrict__ Cbf, long ldcb,
    int K, int op)
{
  const int lane = threadIdx.x & 31;
  const int wave = threadIdx.x >> 5;
  const int hi   = lane >> 4;
  const int l    = lane & 15;
  const int n0   = blockIdx.x * (16 * NB);
  const int m0   = blockIdx.y * 128 + wave * 16;

  const bf16* arow = A + (size_t)(m0 + l) * lda + (hi ? 16 : 0);
  const bf16* wrow[NB];
#pragma unroll
  for (int nb = 0; nb < NB; ++nb)
    wrow[nb] = W + (size_t)(n0 + nb * 16 + l) * K + (hi ? 16 : 0);

  v8f acc[NB];
#pragma unroll
  for (int nb = 0; nb < NB; ++nb)
    acc[nb] = (v8f){0.f, 0.f, 0.f, 0.f, 0.f, 0.f, 0.f, 0.f};

  for (int k0 = 0; k0 < K; k0 += 32) {
    v8bf a0 = *(const v8bf*)(arow + k0);
    v8bf a1 = *(const v8bf*)(arow + k0 + 8);
    v16bf a = __builtin_shufflevector(a0, a1,
                                      0,1,2,3,4,5,6,7,8,9,10,11,12,13,14,15);
#pragma unroll
    for (int nb = 0; nb < NB; ++nb) {
      // speculative near-cache prefetch of the weight stream (global_prefetch_b8)
      __builtin_prefetch(wrow[nb] + k0 + 256, 0, 3);
      v8bf b0 = *(const v8bf*)(wrow[nb] + k0);
      v8bf b1 = *(const v8bf*)(wrow[nb] + k0 + 8);
      v16bf b = __builtin_shufflevector(b0, b1,
                                        0,1,2,3,4,5,6,7,8,9,10,11,12,13,14,15);
      acc[nb] = __builtin_amdgcn_wmma_f32_16x16x32_bf16(false, a, false, b, (short)0,
                                                        acc[nb], false, false);
    }
  }

  const int rbase = m0 + (hi ? 8 : 0);
#pragma unroll
  for (int nb = 0; nb < NB; ++nb) {
    const int   nc = n0 + nb * 16 + l;
    const float bv = bias ? bias[nc] : 0.0f;
#pragma unroll
    for (int i = 0; i < 8; ++i) {
      float v = acc[nb][i] + bv;
      if (op == 1) v = tanhf(v);
      const size_t r = (size_t)(rbase + i);
      C[r * ldc + nc] = v;
      if (Cbf) Cbf[r * ldcb + nc] = (bf16)v;
    }
  }
}

// ---------------------------------------------------------------------------
// GRU gate: elementwise gating + pack_padded masking + output scatter.
// mode 0: forward scan (out row = t*B+b)
// mode 1: reversed scan (out row = clip(len-1-t)*B+b, written only when valid)
// mode 2: decoder (no mask, out row = b)
// grid = (B, H/256), block = 256.
// ---------------------------------------------------------------------------
__global__ void k_gru_gate(
    const float* __restrict__ Gx,     // rows of 3H; row index depends on mode
    const float* __restrict__ Gh,     // [B,3H]
    float* __restrict__ hF, bf16* __restrict__ hBf,
    const int* __restrict__ lengths,  // null => unmasked
    int t, int mode,
    float* __restrict__ outF, bf16* __restrict__ outBf,
    long ostride, int ocol, int accum)
{
  const int  b   = blockIdx.x;
  const int  j   = blockIdx.y * blockDim.x + threadIdx.x;
  const int  len = lengths ? lengths[b] : (t + 1);
  const bool m   = t < len;

  long gxrow;
  if (mode == 0) {
    gxrow = (long)t * B_ + b;
  } else if (mode == 1) {
    int rc = len - 1 - t;
    rc = rc < 0 ? 0 : (rc > T_ - 1 ? T_ - 1 : rc);
    gxrow = (long)rc * B_ + b;
  } else {
    gxrow = b;
  }
  const long orow = gxrow;

  const size_t gx0 = (size_t)gxrow * H3_;
  const size_t gh0 = (size_t)b * H3_;
  const float r = sigmoid_(Gx[gx0 + j]          + Gh[gh0 + j]);
  const float z = sigmoid_(Gx[gx0 + H_ + j]     + Gh[gh0 + H_ + j]);
  const float n = tanhf(Gx[gx0 + 2 * H_ + j] + r * Gh[gh0 + 2 * H_ + j]);

  const size_t hidx = (size_t)b * H_ + j;
  const float  hp   = hF[hidx];
  const float  hn   = (1.0f - z) * n + z * hp;
  const float  hout = m ? hn : hp;
  hF[hidx]  = hout;
  hBf[hidx] = (bf16)hout;

  const size_t oo = (size_t)orow * ostride + ocol + j;
  if (mode == 1) {
    if (m) {                                   // padded slots stay zero (pre-zeroed)
      if (outF)  outF[oo]  = accum ? (outF[oo] + hn) : hn;
      if (outBf) outBf[oo] = (bf16)hn;
    }
  } else {
    const float o = m ? hn : 0.0f;
    if (outF)  outF[oo]  = accum ? (outF[oo] + o) : o;
    if (outBf) outBf[oo] = (bf16)o;
  }
}

// ---------------------------------------------------------------------------
// Embedding gathers (f32 table -> bf16 rows ready for WMMA)
// ---------------------------------------------------------------------------
__global__ void k_enc_embed(const float* __restrict__ tab, const int* __restrict__ ids,
                            bf16* __restrict__ dst) {
  const int row = blockIdx.x;          // row = t*B + b
  const int t = row / B_, b = row % B_;
  const int id = ids[b * T_ + t];
  const float* s = tab + (size_t)id * H_;
  bf16* d = dst + (size_t)row * H_;
  for (int j = threadIdx.x; j < H_; j += blockDim.x) d[j] = (bf16)s[j];
}

__global__ void k_dec_embed(const float* __restrict__ tab, const int* __restrict__ tags,
                            bf16* __restrict__ dst, int t) {
  const int b = blockIdx.x;
  const int tok = (t == 0) ? OID_ : tags[b * T_ + (t - 1)];
  const float* s = tab + (size_t)tok * H_;
  bf16* d = dst + (size_t)b * H_;
  for (int j = threadIdx.x; j < H_; j += blockDim.x) d[j] = (bf16)s[j];
}

// ---------------------------------------------------------------------------
// Attention: scores[b,t] = dot(rnn[b,:], enc[t,b,:])  (grid=(B,T), block=256)
// ---------------------------------------------------------------------------
__global__ void k_scores(const float* __restrict__ rnn, const float* __restrict__ enc,
                         float* __restrict__ scores) {
  __shared__ float red[256];
  const int b = blockIdx.x, t = blockIdx.y;
  const float* x = rnn + (size_t)b * H_;
  const float* e = enc + ((size_t)t * B_ + b) * H_;
  float s = 0.f;
  for (int j = threadIdx.x; j < H_; j += 256) s += x[j] * e[j];
  red[threadIdx.x] = s;
  __syncthreads();
  for (int w = 128; w > 0; w >>= 1) {
    if (threadIdx.x < w) red[threadIdx.x] += red[threadIdx.x + w];
    __syncthreads();
  }
  if (threadIdx.x == 0) scores[(size_t)b * T_ + t] = red[0];
}

// softmax over T per batch row (grid=B, block=64)
__global__ void k_softmax(const float* __restrict__ scores, float* __restrict__ attn) {
  __shared__ float red[64];
  const int b = blockIdx.x, t = threadIdx.x;
  const float v = (t < T_) ? scores[(size_t)b * T_ + t] : -3.402823e38f;
  red[t] = v;
  __syncthreads();
  for (int w = 32; w > 0; w >>= 1) {
    if (t < w) red[t] = fmaxf(red[t], red[t + w]);
    __syncthreads();
  }
  const float mx = red[0];
  __syncthreads();
  const float e = (t < T_) ? __expf(v - mx) : 0.0f;
  red[t] = e;
  __syncthreads();
  for (int w = 32; w > 0; w >>= 1) {
    if (t < w) red[t] += red[t + w];
    __syncthreads();
  }
  if (t < T_) attn[(size_t)b * T_ + t] = e / red[0];
}

// ctx[b,:] = sum_t attn[b,t]*enc[t,b,:] -> bf16 into cat[:, H:2H]
__global__ void k_ctx(const float* __restrict__ attn, const float* __restrict__ enc,
                      bf16* __restrict__ cat) {
  const int b = blockIdx.x;
  const int j = blockIdx.y * blockDim.x + threadIdx.x;
  float s = 0.f;
  for (int t = 0; t < T_; ++t)
    s += attn[(size_t)b * T_ + t] * enc[((size_t)t * B_ + b) * H_ + j];
  cat[(size_t)b * H2_ + H_ + j] = (bf16)s;
}

// masked NLL for step t (logits live in d_out); grid=B, block=NT=256
__global__ void k_nll(const float* __restrict__ logits, const int* __restrict__ tags,
                      const int* __restrict__ lengths, float* __restrict__ nll, int t) {
  __shared__ float red[256];
  const int b = blockIdx.x, i = threadIdx.x;
  const float x = logits[(size_t)b * T_ * NT_ + (size_t)t * NT_ + i];
  red[i] = x;
  __syncthreads();
  for (int w = 128; w > 0; w >>= 1) {
    if (i < w) red[i] = fmaxf(red[i], red[i + w]);
    __syncthreads();
  }
  const float mx = red[0];
  __syncthreads();
  const float e = __expf(x - mx);
  red[i] = e;
  __syncthreads();
  for (int w = 128; w > 0; w >>= 1) {
    if (i < w) red[i] += red[i + w];
    __syncthreads();
  }
  const float lse = mx + __logf(red[0]);
  const int tag = tags[b * T_ + t];
  if (i == tag) {
    const bool m = t < lengths[b];
    nll[(size_t)t * B_ + b] = m ? (lse - x) : 0.0f;
  }
}

// deterministic final reduction: loss = sum(nll)/sum(lengths) (single block)
__global__ void k_loss(const float* __restrict__ nll, const int* __restrict__ lengths,
                       float* __restrict__ out) {
  __shared__ float red[256];
  float s = 0.f;
  for (int i = threadIdx.x; i < T_ * B_; i += 256) s += nll[i];
  red[threadIdx.x] = s;
  __syncthreads();
  for (int w = 128; w > 0; w >>= 1) {
    if (threadIdx.x < w) red[threadIdx.x] += red[threadIdx.x + w];
    __syncthreads();
  }
  if (threadIdx.x == 0) {
    int L = 0;
    for (int b = 0; b < B_; ++b) L += lengths[b];
    out[0] = red[0] / (float)L;
  }
}

// ---------------------------------------------------------------------------
// Host-side launch helpers
// ---------------------------------------------------------------------------
static inline void cast_w(hipStream_t st, const float* s, bf16* d, size_t n) {
  int g = (int)((n + 255) / 256);
  if (g > 4096) g = 4096;
  k_cast_bf16<<<g, 256, 0, st>>>(s, d, n);
}
static inline void zero_ws(hipStream_t st, void* p, size_t bytes) {
  size_t n = bytes / 4;
  int g = (int)((n + 255) / 256);
  if (g > 4096) g = 4096;
  k_zero_u32<<<g, 256, 0, st>>>((uint32_t*)p, n);
}
static inline void gemm(hipStream_t st, const bf16* A, int lda, const bf16* W,
                        const float* bias, float* C, long ldc, bf16* Cbf, long ldcb,
                        int M, int N, int K, int op) {
  if (M >= 1024 && (N % 64) == 0) {
    // big batched GEMMs: maximize A-fragment reuse (4 N-tiles per wave)
    dim3 g(N / 64, M / 128);
    k_gemm_bf16<4><<<g, 256, 0, st>>>(A, lda, W, bias, C, ldc, Cbf, ldcb, K, op);
  } else {
    // small per-step GEMMs: favor wave count (occupancy) over reuse
    dim3 g(N / 32, M / 128);
    k_gemm_bf16<2><<<g, 256, 0, st>>>(A, lda, W, bias, C, ldc, Cbf, ldcb, K, op);
  }
}
static inline void gate(hipStream_t st, const float* Gx, const float* Gh, float* hF,
                        bf16* hBf, const int* lens, int t, int mode, float* outF,
                        bf16* outBf, long ostride, int ocol, int accum) {
  dim3 g(B_, H_ / 256);
  k_gru_gate<<<g, 256, 0, st>>>(Gx, Gh, hF, hBf, lens, t, mode, outF, outBf, ostride,
                                ocol, accum);
}

extern "C" void kernel_launch(void* const* d_in, const int* in_sizes, int n_in,
                              void* d_out, int out_size, void* d_ws, size_t ws_size,
                              hipStream_t stream) {
  (void)in_sizes; (void)n_in; (void)out_size; (void)ws_size;

  const float* enc_embed = (const float*)d_in[0];
  const float* e0_wih    = (const float*)d_in[1];
  const float* e0_whh    = (const float*)d_in[2];
  const float* e0_bih    = (const float*)d_in[3];
  const float* e0_bhh    = (const float*)d_in[4];
  const float* e1_wih    = (const float*)d_in[5];
  const float* e1_whh    = (const float*)d_in[6];
  const float* e1_bih    = (const float*)d_in[7];
  const float* e1_bhh    = (const float*)d_in[8];
  const float* dec_embed = (const float*)d_in[9];
  const float* d_wih     = (const float*)d_in[10];
  const float* d_whh     = (const float*)d_in[11];
  const float* d_bih     = (const float*)d_in[12];
  const float* d_bhh     = (const float*)d_in[13];
  const float* concat_w  = (const float*)d_in[14];
  const float* concat_b  = (const float*)d_in[15];
  const float* out_w     = (const float*)d_in[16];
  const float* out_b     = (const float*)d_in[17];
  const int*   input_ids = (const int*)d_in[18];
  const int*   tag_ids   = (const int*)d_in[19];
  const int*   lengths   = (const int*)d_in[20];
  float*       out       = (float*)d_out;

  // ---- bump allocator over d_ws ----
  char*  base = (char*)d_ws;
  size_t off  = 0;
  auto alloc = [&](size_t bytes) -> void* {
    void* p = base + off;
    off = (off + bytes + 255) & ~(size_t)255;
    return p;
  };

  // bf16 weight mirrors (L2-resident working set)
  bf16* wb_e0_wih = (bf16*)alloc((size_t)2 * H3_ * H_  * 2);
  bf16* wb_e0_whh = (bf16*)alloc((size_t)2 * H3_ * H_  * 2);
  bf16* wb_e1_wih = (bf16*)alloc((size_t)2 * H3_ * H2_ * 2);
  bf16* wb_e1_whh = (bf16*)alloc((size_t)2 * H3_ * H_  * 2);
  bf16* wb_d_wih  = (bf16*)alloc((size_t)2 * H3_ * H_  * 2);
  bf16* wb_d_whh  = (bf16*)alloc((size_t)2 * H3_ * H_  * 2);
  bf16* wb_cc     = (bf16*)alloc((size_t)H_ * H2_ * 2);
  bf16* wb_out    = (bf16*)alloc((size_t)NT_ * H_ * 2);
  // bf16 activations
  bf16* emb_bf    = (bf16*)alloc((size_t)T_ * B_ * H_  * 2);
  bf16* x1_bf     = (bf16*)alloc((size_t)T_ * B_ * H2_ * 2);
  bf16* x_bf      = (bf16*)alloc((size_t)B_ * H_  * 2);
  bf16* cat_bf    = (bf16*)alloc((size_t)B_ * H2_ * 2);
  bf16* cc_bf     = (bf16*)alloc((size_t)B_ * H_  * 2);
  bf16* h_l0f_bf  = (bf16*)alloc((size_t)B_ * H_ * 2);
  bf16* h_l0b_bf  = (bf16*)alloc((size_t)B_ * H_ * 2);
  bf16* h_l1f_bf  = (bf16*)alloc((size_t)B_ * H_ * 2);
  bf16* h_l1b_bf  = (bf16*)alloc((size_t)B_ * H_ * 2);
  // f32 buffers
  float* Gx      = (float*)alloc((size_t)T_ * B_ * H3_ * 4);
  float* Ghd     = (float*)alloc((size_t)B_ * H3_ * 4);
  float* Gxd     = (float*)alloc((size_t)B_ * H3_ * 4);
  float* h_l0f   = (float*)alloc((size_t)B_ * H_ * 4);
  float* h_l0b   = (float*)alloc((size_t)B_ * H_ * 4);
  float* h_l1f   = (float*)alloc((size_t)B_ * H_ * 4);
  float* h_l1b   = (float*)alloc((size_t)B_ * H_ * 4);
  float* enc_out = (float*)alloc((size_t)T_ * B_ * H_ * 4);
  float* ccF     = (float*)alloc((size_t)B_ * H_ * 4);
  float* scores  = (float*)alloc((size_t)B_ * T_ * 4);
  float* attn    = (float*)alloc((size_t)B_ * T_ * 4);
  float* nll     = (float*)alloc((size_t)T_ * B_ * 4);

  // ---- weight conversion (once per call; deterministic) ----
  cast_w(stream, e0_wih,   wb_e0_wih, (size_t)2 * H3_ * H_);
  cast_w(stream, e0_whh,   wb_e0_whh, (size_t)2 * H3_ * H_);
  cast_w(stream, e1_wih,   wb_e1_wih, (size_t)2 * H3_ * H2_);
  cast_w(stream, e1_whh,   wb_e1_whh, (size_t)2 * H3_ * H_);
  cast_w(stream, d_wih,    wb_d_wih,  (size_t)2 * H3_ * H_);
  cast_w(stream, d_whh,    wb_d_whh,  (size_t)2 * H3_ * H_);
  cast_w(stream, concat_w, wb_cc,     (size_t)H_ * H2_);
  cast_w(stream, out_w,    wb_out,    (size_t)NT_ * H_);

  // ---- encoder embedding gather + zero-init scatter targets ----
  k_enc_embed<<<T_ * B_, 256, 0, stream>>>(enc_embed, input_ids, emb_bf);
  zero_ws(stream, x1_bf, (size_t)T_ * B_ * H2_ * 2);  // bwd halves rely on zeros

  // ======================= encoder layer 0 =======================
  // forward direction
  gemm(stream, emb_bf, H_, wb_e0_wih, e0_bih, Gx, H3_, nullptr, 0,
       T_ * B_, H3_, H_, 0);
  zero_ws(stream, h_l0f, (size_t)B_ * H_ * 4);
  zero_ws(stream, h_l0f_bf, (size_t)B_ * H_ * 2);
  for (int t = 0; t < T_; ++t) {
    gemm(stream, h_l0f_bf, H_, wb_e0_whh, e0_bhh, Ghd, H3_, nullptr, 0,
         B_, H3_, H_, 0);
    gate(stream, Gx, Ghd, h_l0f, h_l0f_bf, lengths, t, 0,
         nullptr, x1_bf, H2_, 0, 0);
  }
  // backward direction
  gemm(stream, emb_bf, H_, wb_e0_wih + (size_t)H3_ * H_, e0_bih + H3_,
       Gx, H3_, nullptr, 0, T_ * B_, H3_, H_, 0);
  zero_ws(stream, h_l0b, (size_t)B_ * H_ * 4);
  zero_ws(stream, h_l0b_bf, (size_t)B_ * H_ * 2);
  for (int t = 0; t < T_; ++t) {
    gemm(stream, h_l0b_bf, H_, wb_e0_whh + (size_t)H3_ * H_, e0_bhh + H3_,
         Ghd, H3_, nullptr, 0, B_, H3_, H_, 0);
    gate(stream, Gx, Ghd, h_l0b, h_l0b_bf, lengths, t, 1,
         nullptr, x1_bf, H2_, H_, 0);
  }

  // ======================= encoder layer 1 =======================
  // forward: writes of1 into enc_out
  gemm(stream, x1_bf, H2_, wb_e1_wih, e1_bih, Gx, H3_, nullptr, 0,
       T_ * B_, H3_, H2_, 0);
  zero_ws(stream, h_l1f, (size_t)B_ * H_ * 4);
  zero_ws(stream, h_l1f_bf, (size_t)B_ * H_ * 2);
  for (int t = 0; t < T_; ++t) {
    gemm(stream, h_l1f_bf, H_, wb_e1_whh, e1_bhh, Ghd, H3_, nullptr, 0,
         B_, H3_, H_, 0);
    gate(stream, Gx, Ghd, h_l1f, h_l1f_bf, lengths, t, 0,
         enc_out, nullptr, H_, 0, 0);
  }
  // backward: accumulates ob1 into enc_out (enc_out = of1 + ob1)
  gemm(stream, x1_bf, H2_, wb_e1_wih + (size_t)H3_ * H2_, e1_bih + H3_,
       Gx, H3_, nullptr, 0, T_ * B_, H3_, H2_, 0);
  zero_ws(stream, h_l1b, (size_t)B_ * H_ * 4);
  zero_ws(stream, h_l1b_bf, (size_t)B_ * H_ * 2);
  for (int t = 0; t < T_; ++t) {
    gemm(stream, h_l1b_bf, H_, wb_e1_whh + (size_t)H3_ * H_, e1_bhh + H3_,
         Ghd, H3_, nullptr, 0, B_, H3_, H_, 0);
    gate(stream, Gx, Ghd, h_l1b, h_l1b_bf, lengths, t, 1,
         enc_out, nullptr, H_, 0, 1);
  }

  // ======================= decoder =======================
  // decoder hidden states start as encoder layer-0 final states (reuse buffers)
  float* hd0 = h_l0f; bf16* hd0_bf = h_l0f_bf;
  float* hd1 = h_l0b; bf16* hd1_bf = h_l0b_bf;

  for (int t = 0; t < T_; ++t) {
    k_dec_embed<<<B_, 256, 0, stream>>>(dec_embed, tag_ids, x_bf, t);

    // decoder GRU layer 0: hd0 = gru(x, hd0)
    gemm(stream, x_bf, H_, wb_d_wih, d_bih, Gxd, H3_, nullptr, 0, B_, H3_, H_, 0);
    gemm(stream, hd0_bf, H_, wb_d_whh, d_bhh, Ghd, H3_, nullptr, 0, B_, H3_, H_, 0);
    gate(stream, Gxd, Ghd, hd0, hd0_bf, nullptr, t, 2, nullptr, nullptr, 1, 0, 0);

    // decoder GRU layer 1: hd1 = gru(hd0, hd1); rnn_out -> cat[:, 0:H] (bf16)
    gemm(stream, hd0_bf, H_, wb_d_wih + (size_t)H3_ * H_, d_bih + H3_,
         Gxd, H3_, nullptr, 0, B_, H3_, H_, 0);
    gemm(stream, hd1_bf, H_, wb_d_whh + (size_t)H3_ * H_, d_bhh + H3_,
         Ghd, H3_, nullptr, 0, B_, H3_, H_, 0);
    gate(stream, Gxd, Ghd, hd1, hd1_bf, nullptr, t, 2, nullptr, cat_bf, H2_, 0, 0);

    // attention over encoder outputs
    k_scores<<<dim3(B_, T_), 256, 0, stream>>>(hd1, enc_out, scores);
    k_softmax<<<B_, 64, 0, stream>>>(scores, attn);
    k_ctx<<<dim3(B_, H_ / 256), 256, 0, stream>>>(attn, enc_out, cat_bf);

    // cc = tanh([rnn_out, ctx] @ concat_w^T + concat_b)   (fused tanh epilogue)
    gemm(stream, cat_bf, H2_, wb_cc, concat_b, ccF, H_, cc_bf, H_, B_, H_, H2_, 1);

    // logits -> d_out[b, t, :]
    gemm(stream, cc_bf, H_, wb_out, out_b, out + (size_t)t * NT_,
         (long)T_ * NT_, nullptr, 0, B_, NT_, H_, 0);

    // masked NLL for this step
    k_nll<<<B_, 256, 0, stream>>>(out, tag_ids, lengths, nll, t);
  }

  // loss scalar at d_out[B*T*NT]
  k_loss<<<1, 256, 0, stream>>>(nll, lengths, out + (size_t)B_ * T_ * NT_);
}
